// SimpleSAE_75780402971103
// MI455X (gfx1250) — matrix operand order repaired and checked by reading, not compiled, not run on hardware
//
#include <hip/hip_runtime.h>
#include <hip/hip_bf16.h>
#include <stdint.h>

#define BATCH   4096
#define IN_DIM  1024
#define HID     16384
#define KACT    64

typedef __attribute__((ext_vector_type(16))) __bf16 v16bf;
typedef __attribute__((ext_vector_type(8)))  float  v8f;

__device__ __forceinline__ unsigned short f32_to_bf16(float f) {
    uint32_t b = __float_as_uint(f);
    uint32_t r = b + 0x7FFFu + ((b >> 16) & 1u);   // round to nearest even
    return (unsigned short)(r >> 16);
}

__device__ __forceinline__ v16bf load_frag16(const unsigned short* lds, int u0, int u1) {
    union { v16bf v; uint4 q[2]; } u;
    u.q[0] = *(const uint4*)(lds + u0);   // 16B chunk (VGPRs 0..3)
    u.q[1] = *(const uint4*)(lds + u1);   // 16B chunk (VGPRs 4..7)
    return u.v;
}

// ---------------------------------------------------------------------------
// Kernel 1: encoded = x @ W_enc^T  via V_WMMA_F32_16X16X32_BF16
// Block: 256 thr = 8 waves, tile M=64 x N=256, K-step 32, double-buffered LDS.
// Per wave: 32M x 64N -> 8 WMMA per K-step, 12 ds_load_b128 (1.5 loads/WMMA).
// LDS rows padded to 80B (40 ushorts) -> conflict-free b128 A-fragment loads.
// ---------------------------------------------------------------------------
__global__ void __launch_bounds__(256) sae_encoder_wmma(
    const float* __restrict__ x, const float* __restrict__ W_enc,
    float* __restrict__ enc_out)
{
    __shared__ __align__(16) unsigned short As[2][64 * 40];    // 2 x 5 KB
    __shared__ __align__(16) unsigned short Bs[2][256 * 40];   // 2 x 20 KB

    const int tid   = threadIdx.x;
    const int mBase = blockIdx.y * 64;
    const int nBase = blockIdx.x * 256;

    const int wave = tid >> 5;
    const int lane = tid & 31;
    const int half = lane >> 4;       // 0: lanes 0-15, 1: lanes 16-31
    const int l16  = lane & 15;
    const int mw   = wave >> 2;       // 0..1 -> M offset 0/32
    const int nw   = wave & 3;        // 0..3 -> N offset 0/64/128/192

    v8f acc[2][4] = {};

    const int srow = tid >> 3;            // 0..31
    const int scol = (tid & 7) * 4;       // 0..28
    const float* xb = x     + (size_t)(mBase + srow) * IN_DIM + scol;
    const float* wb = W_enc + (size_t)(nBase + srow) * IN_DIM + scol;

    float4 xr[2], wr[8];                  // staged next tile (registers)

    auto load_next = [&](int k0) {
        #pragma unroll
        for (int r = 0; r < 2; ++r) xr[r] = *(const float4*)(xb + (size_t)r * 32 * IN_DIM + k0);
        #pragma unroll
        for (int r = 0; r < 8; ++r) wr[r] = *(const float4*)(wb + (size_t)r * 32 * IN_DIM + k0);
    };
    auto store_stage = [&](int buf) {
        #pragma unroll
        for (int r = 0; r < 2; ++r) {
            union { uint2 u2; unsigned short s[4]; } p;
            p.s[0] = f32_to_bf16(xr[r].x); p.s[1] = f32_to_bf16(xr[r].y);
            p.s[2] = f32_to_bf16(xr[r].z); p.s[3] = f32_to_bf16(xr[r].w);
            *(uint2*)(&As[buf][(r * 32 + srow) * 40 + scol]) = p.u2;
        }
        #pragma unroll
        for (int r = 0; r < 8; ++r) {
            union { uint2 u2; unsigned short s[4]; } p;
            p.s[0] = f32_to_bf16(wr[r].x); p.s[1] = f32_to_bf16(wr[r].y);
            p.s[2] = f32_to_bf16(wr[r].z); p.s[3] = f32_to_bf16(wr[r].w);
            *(uint2*)(&Bs[buf][(r * 32 + srow) * 40 + scol]) = p.u2;
        }
    };

    // prologue: stage K-tile 0 into buffer 0
    load_next(0);
    store_stage(0);
    __syncthreads();

    int cur = 0;
    for (int k0 = 0; k0 < IN_DIM; k0 += 32) {
        const bool more = (k0 + 32) < IN_DIM;
        if (more) {
            load_next(k0 + 32);                                   // overlap with WMMA
            if (k0 + 64 < IN_DIM)
                __builtin_prefetch(wb + k0 + 96, 0, 1);           // global_prefetch_b8
        }

        // A fragments: lane<16 holds K 0..7/16..23, lane>=16 holds K 8..15/24..31
        v16bf afrag[2];
        #pragma unroll
        for (int ma = 0; ma < 2; ++ma) {
            const int aoff = (mw * 32 + ma * 16 + l16) * 40 + half * 8;
            afrag[ma] = load_frag16(&As[cur][0], aoff, aoff + 16);
        }
        #pragma unroll
        for (int t = 0; t < 4; ++t) {
            // B fragment: lane<16 holds K 0..7/16..23, lane>=16 holds K 8..15/24..31 of column n
            const int boff = (nw * 64 + t * 16 + l16) * 40 + half * 8;
            v16bf bfrag = load_frag16(&Bs[cur][0], boff, boff + 16);
            #pragma unroll
            for (int ma = 0; ma < 2; ++ma)
                acc[ma][t] = __builtin_amdgcn_wmma_f32_16x16x32_bf16(
                    false, afrag[ma], false, bfrag, (short)0, acc[ma][t], false, false);
        }

        if (more) store_stage(cur ^ 1);
        __syncthreads();
        cur ^= 1;
    }

    // C layout: VGPR r = row (half*8 + r), lane&15 = col
    #pragma unroll
    for (int ma = 0; ma < 2; ++ma) {
        #pragma unroll
        for (int t = 0; t < 4; ++t) {
            const int colg = nBase + nw * 64 + t * 16 + l16;
            #pragma unroll
            for (int r = 0; r < 8; ++r) {
                const int rowg = mBase + mw * 32 + ma * 16 + half * 8 + r;
                enc_out[(size_t)rowg * HID + colg] = acc[ma][t][r];
            }
        }
    }
}

// ---------------------------------------------------------------------------
// Kernel 2: exact top-64 per row. Values live in registers (64/thread).
// Radix-select (4x8-bit) on order-preserving uint keys; deterministic
// h-ordered compaction via per-wave ballots + serial prefix (no nondeterminism
// across replays).
// ---------------------------------------------------------------------------
__global__ void __launch_bounds__(256) sae_topk(
    float* __restrict__ enc, float* __restrict__ vals, int* __restrict__ idxs,
    int doCompact)
{
    __shared__ uint32_t hist[256];
    __shared__ uint32_t wmask[64 * 8];
    __shared__ uint32_t wbase[64 * 8];
    __shared__ uint32_t s_bin, s_krem;

    const int tid  = threadIdx.x;
    const int b    = blockIdx.x;
    const int w    = tid >> 5;
    const int lane = tid & 31;
    const uint32_t lmask = (1u << lane) - 1u;
    float* row = enc + (size_t)b * HID;

    uint32_t u[64];
    #pragma unroll
    for (int j = 0; j < 64; ++j) {
        uint32_t bb = __float_as_uint(row[j * 256 + tid]);
        u[j] = (bb & 0x80000000u) ? ~bb : (bb | 0x80000000u);
    }

    if (tid == 0) s_krem = KACT;
    uint32_t prefix = 0;
    for (int round = 0; round < 4; ++round) {
        hist[tid] = 0;
        __syncthreads();
        const int shift = 24 - 8 * round;
        #pragma unroll
        for (int j = 0; j < 64; ++j) {
            uint32_t v = u[j];
            bool match = (round == 0) || ((v >> (shift + 8)) == prefix);
            if (match) atomicAdd(&hist[(v >> shift) & 255u], 1u);
        }
        __syncthreads();
        if (tid == 0) {
            uint32_t kr = s_krem, accc = 0;
            int bsel = 0;
            for (int bin = 255; bin >= 0; --bin) {
                accc += hist[bin];
                if (accc >= kr) { bsel = bin; s_krem = kr - (accc - hist[bin]); break; }
            }
            s_bin = (uint32_t)bsel;
        }
        __syncthreads();
        prefix = (prefix << 8) | s_bin;
        __syncthreads();
    }
    const uint32_t tval = prefix;       // exact bits of the 64th-largest key
    const uint32_t krem = s_krem;       // #ties at tval to keep (h-order first)

    // rank ties deterministically, build final selection mask
    unsigned long long fsel = 0ull;
    #pragma unroll
    for (int j = 0; j < 64; ++j) {
        bool s = (u[j] == tval);
        unsigned long long bal = __ballot(s);
        if (lane == 0) wmask[j * 8 + w] = (uint32_t)bal;
        if (u[j] > tval) fsel |= (1ull << j);
    }
    __syncthreads();
    if (tid == 0) {
        uint32_t run = 0;
        for (int q = 0; q < 512; ++q) { wbase[q] = run; run += __popc(wmask[q]); }
    }
    __syncthreads();
    #pragma unroll
    for (int j = 0; j < 64; ++j) {
        if (u[j] == tval) {
            uint32_t rank = wbase[j * 8 + w] + __popc(wmask[j * 8 + w] & lmask);
            if (rank < krem) fsel |= (1ull << j);
        }
    }
    __syncthreads();

    // rank selected elements in h order -> compaction slots 0..63
    #pragma unroll
    for (int j = 0; j < 64; ++j) {
        bool s = (fsel >> j) & 1ull;
        unsigned long long bal = __ballot(s);
        if (lane == 0) wmask[j * 8 + w] = (uint32_t)bal;
    }
    __syncthreads();
    if (tid == 0) {
        uint32_t run = 0;
        for (int q = 0; q < 512; ++q) { wbase[q] = run; run += __popc(wmask[q]); }
    }
    __syncthreads();
    #pragma unroll
    for (int j = 0; j < 64; ++j) {
        bool s = (fsel >> j) & 1ull;
        uint32_t v = u[j];
        uint32_t bb = (v & 0x80000000u) ? (v & 0x7FFFFFFFu) : ~v;   // undo key transform
        float f = __uint_as_float(bb);
        if (s && doCompact) {
            uint32_t slot = wbase[j * 8 + w] + __popc(wmask[j * 8 + w] & lmask);
            vals[(size_t)b * KACT + slot] = f;
            idxs[(size_t)b * KACT + slot] = j * 256 + tid;
        }
        row[j * 256 + tid] = s ? f : 0.0f;   // dense sparse_encoded output
    }
}

// ---------------------------------------------------------------------------
// Kernel 3: W_dec [1024 x 16384] -> W_decT [16384 x 1024] (coalesced decoder)
// ---------------------------------------------------------------------------
__global__ void __launch_bounds__(256) sae_transpose(
    const float* __restrict__ W, float* __restrict__ WT)
{
    __shared__ float tile[32][33];
    const int h0 = blockIdx.x * 32;
    const int i0 = blockIdx.y * 32;
    const int tx = threadIdx.x, ty = threadIdx.y;
    #pragma unroll
    for (int r = 0; r < 4; ++r)
        tile[ty + r * 8][tx] = W[(size_t)(i0 + ty + r * 8) * HID + (h0 + tx)];
    __syncthreads();
    #pragma unroll
    for (int r = 0; r < 4; ++r)
        WT[(size_t)(h0 + ty + r * 8) * IN_DIM + (i0 + tx)] = tile[tx][ty + r * 8];
}

// ---------------------------------------------------------------------------
// Kernel 4: decoded[b,:] = sum_k val_k * W_decT[h_k,:]   (coalesced)
// ---------------------------------------------------------------------------
__global__ void __launch_bounds__(256) sae_decode_T(
    const float* __restrict__ vals, const int* __restrict__ idxs,
    const float* __restrict__ WT, float* __restrict__ out)
{
    __shared__ float sv[KACT];
    __shared__ int   sh[KACT];
    const int tid = threadIdx.x, b = blockIdx.x;
    if (tid < KACT) { sv[tid] = vals[(size_t)b * KACT + tid];
                      sh[tid] = idxs[(size_t)b * KACT + tid]; }
    __syncthreads();
    float a0 = 0.f, a1 = 0.f, a2 = 0.f, a3 = 0.f;
    for (int k = 0; k < KACT; ++k) {
        float v = sv[k];
        const float* wrp = WT + (size_t)sh[k] * IN_DIM;
        a0 += v * wrp[tid];       a1 += v * wrp[tid + 256];
        a2 += v * wrp[tid + 512]; a3 += v * wrp[tid + 768];
    }
    float* o = out + (size_t)b * IN_DIM;
    o[tid] = a0; o[tid + 256] = a1; o[tid + 512] = a2; o[tid + 768] = a3;
}

// Fallback decoder: gather straight from W_dec (L2-resident, uncoalesced)
__global__ void __launch_bounds__(256) sae_decode_direct(
    const float* __restrict__ vals, const int* __restrict__ idxs,
    const float* __restrict__ Wd, float* __restrict__ out)
{
    __shared__ float sv[KACT];
    __shared__ int   sh[KACT];
    const int tid = threadIdx.x, b = blockIdx.x;
    if (tid < KACT) { sv[tid] = vals[(size_t)b * KACT + tid];
                      sh[tid] = idxs[(size_t)b * KACT + tid]; }
    __syncthreads();
    float a0 = 0.f, a1 = 0.f, a2 = 0.f, a3 = 0.f;
    for (int k = 0; k < KACT; ++k) {
        float v = sv[k]; int h = sh[k];
        a0 += v * Wd[(size_t)(tid)       * HID + h];
        a1 += v * Wd[(size_t)(tid + 256) * HID + h];
        a2 += v * Wd[(size_t)(tid + 512) * HID + h];
        a3 += v * Wd[(size_t)(tid + 768) * HID + h];
    }
    float* o = out + (size_t)b * IN_DIM;
    o[tid] = a0; o[tid + 256] = a1; o[tid + 512] = a2; o[tid + 768] = a3;
}

// Last-resort decoder: scan the dense sparse row (no workspace needed)
__global__ void __launch_bounds__(256) sae_decode_dense(
    const float* __restrict__ sp, const float* __restrict__ Wd,
    float* __restrict__ out)
{
    __shared__ float sS[256];
    const int tid = threadIdx.x, b = blockIdx.x;
    float a0 = 0.f, a1 = 0.f, a2 = 0.f, a3 = 0.f;
    for (int j = 0; j < 64; ++j) {
        sS[tid] = sp[(size_t)b * HID + j * 256 + tid];
        __syncthreads();
        for (int t = 0; t < 256; ++t) {
            float s = sS[t];
            if (s != 0.f) {
                int h = j * 256 + t;
                a0 += s * Wd[(size_t)(tid)       * HID + h];
                a1 += s * Wd[(size_t)(tid + 256) * HID + h];
                a2 += s * Wd[(size_t)(tid + 512) * HID + h];
                a3 += s * Wd[(size_t)(tid + 768) * HID + h];
            }
        }
        __syncthreads();
    }
    float* o = out + (size_t)b * IN_DIM;
    o[tid] = a0; o[tid + 256] = a1; o[tid + 512] = a2; o[tid + 768] = a3;
}

extern "C" void kernel_launch(void* const* d_in, const int* in_sizes, int n_in,
                              void* d_out, int out_size, void* d_ws, size_t ws_size,
                              hipStream_t stream) {
    (void)in_sizes; (void)n_in; (void)out_size;
    const float* x     = (const float*)d_in[0];
    const float* W_enc = (const float*)d_in[1];
    const float* W_dec = (const float*)d_in[2];

    float* out     = (float*)d_out;
    float* dec_out = out;                               // [4096,1024]
    float* sp_out  = out + (size_t)BATCH * IN_DIM;      // [4096,16384] (also GEMM scratch)

    const size_t needVI = (size_t)BATCH * KACT * (sizeof(float) + sizeof(int)); // 2 MB
    const size_t needT  = needVI + (size_t)HID * IN_DIM * sizeof(float);        // +64 MB
    float* vals = (float*)d_ws;
    int*   idxs = (int*)((char*)d_ws + (size_t)BATCH * KACT * sizeof(float));
    float* WT   = (float*)((char*)d_ws + needVI);
    const bool haveVI = ws_size >= needVI;
    const bool haveT  = ws_size >= needT;

    // 1) dense encoder GEMM (WMMA bf16) into the sparse-output region
    dim3 gE(HID / 256, BATCH / 64);
    sae_encoder_wmma<<<gE, 256, 0, stream>>>(x, W_enc, sp_out);

    // 2) exact top-64 per row; sparsify in place; compact (val,idx) into ws
    sae_topk<<<BATCH, 256, 0, stream>>>(sp_out, vals, idxs, haveVI ? 1 : 0);

    // 3+4) decoder
    if (haveT) {
        sae_transpose<<<dim3(HID / 32, IN_DIM / 32), dim3(32, 8), 0, stream>>>(W_dec, WT);
        sae_decode_T<<<BATCH, 256, 0, stream>>>(vals, idxs, WT, dec_out);
    } else if (haveVI) {
        sae_decode_direct<<<BATCH, 256, 0, stream>>>(vals, idxs, W_dec, dec_out);
    } else {
        sae_decode_dense<<<BATCH, 256, 0, stream>>>(sp_out, W_dec, dec_out);
    }
}
